// AdaptiveAdjacency_14714557956225
// MI455X (gfx1250) — compile-verified, matrix operand order checked by reading, not run on hardware
//
#include <hip/hip_runtime.h>
#include <hip/hip_bf16.h>

typedef __attribute__((ext_vector_type(16))) _Float16 v16h;
typedef __attribute__((ext_vector_type(8)))  float    v8f;

#define NNODES 16384
#define HDIM   128
#define TOPK   20
#define WAVES_PER_BLK 8
#define COLS_PER_WAVE (NNODES / WAVES_PER_BLK)   // 2048
#define TILES_PER_WAVE (COLS_PER_WAVE / 16)      // 128
#define NLISTS (WAVES_PER_BLK * 2)               // 16 lists per row

// ---------------------------------------------------------------------------
// Kernel 1: L2-normalize each row, write f16 copy to workspace.
// One 128-thread block per row.
// ---------------------------------------------------------------------------
__global__ __launch_bounds__(HDIM)
void normalize_rows_f16(const float* __restrict__ emb, _Float16* __restrict__ out) {
    __shared__ float red[HDIM];
    const int row = blockIdx.x;
    const int t   = threadIdx.x;
    const float x = emb[(size_t)row * HDIM + t];
    red[t] = x * x;
    __syncthreads();
    #pragma unroll
    for (int s = HDIM / 2; s > 0; s >>= 1) {
        if (t < s) red[t] += red[t + s];
        __syncthreads();
    }
    const float r = rsqrtf(fmaxf(red[0], 1e-12f));
    out[(size_t)row * HDIM + t] = (_Float16)(x * r);
}

// Contiguous per-lane fragment load. The SAME mapping is used for the A and
// B operands, so the WMMA K-reduction is invariant to the hardware K-slot
// permutation (any shared bijection over K=0..31 yields the correct dot).
__device__ __forceinline__ v16h load_frag(const _Float16* __restrict__ rowPtr,
                                          int half, int chunk) {
    return *(const v16h*)(rowPtr + chunk * 32 + half * 16);
}

// ---------------------------------------------------------------------------
// Kernel 2: fused similarity GEMM (WMMA f16 -> f32) + per-row top-20.
// Block = 16 output rows, 8 waves; wave w sweeps columns [w*2048, (w+1)*2048).
// All 32 lanes participate in selection: lane = (row lane%16, col-half lane/16),
// each lane owns a private sorted top-20 list; 16 lists per row merged at end.
// ---------------------------------------------------------------------------
__global__ __launch_bounds__(256)
void sim_topk_kernel(const _Float16* __restrict__ embH, float* __restrict__ out) {
    __shared__ float stage[WAVES_PER_BLK][16][16];          // [wave][row][col]
    __shared__ float lvals[WAVES_PER_BLK][2][16][TOPK];     // [wave][half][row][k]
    __shared__ int   lidxs[WAVES_PER_BLK][2][16][TOPK];

    const int lane = threadIdx.x & 31;
    const int wave = threadIdx.x >> 5;
    const int lr   = lane & 15;      // row (scan) / row-col within 16-tile (frags)
    const int half = lane >> 4;      // K-half (frags) / column-half (scan)
    const int rowBase = blockIdx.x * 16;

    // Init this lane's private candidate list (exclusively owned).
    float* LV = &lvals[wave][half][lr][0];
    int*   LI = &lidxs[wave][half][lr][0];
    #pragma unroll
    for (int k = 0; k < TOPK; ++k) { LV[k] = -3.0e38f; LI[k] = 0; }
    float vmin = -3.0e38f;           // list minimum, kept register-resident

    // A fragments for this block's 16 rows (resident in registers).
    const _Float16* arow = embH + (size_t)(rowBase + lr) * HDIM;
    const v16h a0 = load_frag(arow, half, 0);
    const v16h a1 = load_frag(arow, half, 1);
    const v16h a2 = load_frag(arow, half, 2);
    const v16h a3 = load_frag(arow, half, 3);

    const int colStart = wave * COLS_PER_WAVE;

    // Pipeline prologue: B fragments for tile 0.
    const _Float16* brow0 = embH + (size_t)(colStart + lr) * HDIM;
    v16h b0 = load_frag(brow0, half, 0);
    v16h b1 = load_frag(brow0, half, 1);
    v16h b2 = load_frag(brow0, half, 2);
    v16h b3 = load_frag(brow0, half, 3);

    #pragma unroll 2                       // ping-pong: renaming kills b=nb copies
    for (int t = 0; t < TILES_PER_WAVE; ++t) {
        const int colBase = colStart + t * 16;

        // Prefetch next tile's B fragments (clamped index -> branch-free).
        const int tn = (t + 1 < TILES_PER_WAVE) ? (t + 1) : t;
        const _Float16* nbrow = embH + (size_t)(colStart + tn * 16 + lr) * HDIM;
        v16h nb0 = load_frag(nbrow, half, 0);
        v16h nb1 = load_frag(nbrow, half, 1);
        v16h nb2 = load_frag(nbrow, half, 2);
        v16h nb3 = load_frag(nbrow, half, 3);

        v8f c = {};
        c = __builtin_amdgcn_wmma_f32_16x16x32_f16(false, a0, false, b0,
                                                   (short)0, c, false, false);
        c = __builtin_amdgcn_wmma_f32_16x16x32_f16(false, a1, false, b1,
                                                   (short)0, c, false, false);
        c = __builtin_amdgcn_wmma_f32_16x16x32_f16(false, a2, false, b2,
                                                   (short)0, c, false, false);
        c = __builtin_amdgcn_wmma_f32_16x16x32_f16(false, a3, false, b3,
                                                   (short)0, c, false, false);

        // Spill accumulator tile: VGPR r holds M = r + 8*half, N = lane%16.
        #pragma unroll
        for (int r = 0; r < 8; ++r)
            stage[wave][r + 8 * half][lr] = c[r];

        __builtin_amdgcn_wave_barrier();   // scheduling fence; LDS in-order per wave

        // Every lane scans 8 sims of its (row, col-half) as two b128 LDS loads.
        {
            const int m = lr;
            const float4 q0 = *(const float4*)&stage[wave][m][half * 8];
            const float4 q1 = *(const float4*)&stage[wave][m][half * 8 + 4];
            float v8[8] = { q0.x, q0.y, q0.z, q0.w, q1.x, q1.y, q1.z, q1.w };

            float mx = fmaxf(fmaxf(fmaxf(v8[0], v8[1]), fmaxf(v8[2], v8[3])),
                             fmaxf(fmaxf(v8[4], v8[5]), fmaxf(v8[6], v8[7])));

            if (mx > vmin) {                       // rare after warm-up
                const int cbase = colBase + half * 8;
                #pragma unroll
                for (int cc = 0; cc < 8; ++cc) {
                    const float v = v8[cc];
                    if (v > vmin) {
                        // New slot-19 value: old LV[18] if it shifts down, else v.
                        vmin = fminf(LV[TOPK - 2], v);
                        int p = TOPK - 1;
                        #pragma clang loop unroll(disable)
                        while (p > 0 && LV[p - 1] < v) {  // strict '<' => stable ties
                            LV[p] = LV[p - 1];
                            LI[p] = LI[p - 1];
                            --p;
                        }
                        LV[p] = v;
                        LI[p] = cbase + cc;
                    }
                }
            }
        }

        b0 = nb0; b1 = nb1; b2 = nb2; b3 = nb3;
    }

    __syncthreads();

    // Merge the 16 per-(wave,half) sorted lists for each row. List order
    // (wave, half): waves are increasing-column, halves interleave 8-col groups; strict '>' =>
    // ties pick the lowest (wave, half), not always the lowest column.
    if (threadIdx.x < 16) {
        const int m   = threadIdx.x;
        const int row = rowBase + m;
        int ptr[NLISTS];
        #pragma unroll
        for (int l = 0; l < NLISTS; ++l) ptr[l] = 0;

        float* outVal = out + (size_t)row * TOPK;
        float* outIdx = out + (size_t)NNODES * TOPK + (size_t)row * TOPK;

        for (int j = 0; j < TOPK; ++j) {
            float best = -3.4e38f;
            int   bl   = 0;
            #pragma unroll
            for (int l = 0; l < NLISTS; ++l) {
                if (ptr[l] < TOPK) {
                    const float v = lvals[l >> 1][l & 1][m][ptr[l]];
                    if (v > best) { best = v; bl = l; }
                }
            }
            outVal[j] = best;
            outIdx[j] = (float)lidxs[bl >> 1][bl & 1][m][ptr[bl]];
            ptr[bl]++;
        }
    }
}

extern "C" void kernel_launch(void* const* d_in, const int* in_sizes, int n_in,
                              void* d_out, int out_size, void* d_ws, size_t ws_size,
                              hipStream_t stream) {
    const float* emb  = (const float*)d_in[0];
    float*       out  = (float*)d_out;
    _Float16*    embH = (_Float16*)d_ws;   // 16384*128*2 = 4 MB scratch

    normalize_rows_f16<<<NNODES, HDIM, 0, stream>>>(emb, embH);
    sim_topk_kernel<<<NNODES / 16, 256, 0, stream>>>(embH, out);
}
